// DGCNMIL_60979945669303
// MI455X (gfx1250) — matrix operand order repaired and checked by reading, not compiled, not run on hardware
//
#include <hip/hip_runtime.h>

typedef __attribute__((ext_vector_type(16))) __bf16 bf16x16;
typedef __attribute__((ext_vector_type(8)))  __bf16 bf16x8;
typedef __attribute__((ext_vector_type(8)))  float  f32x8;

// ---- CDNA5 async global->LDS copy (ASYNCcnt-tracked, bypasses VGPRs) ----
__device__ __forceinline__ void async_copy_b128(const __bf16* g, const __bf16* l) {
  // generic shared pointer: addr[31:0] is the LDS byte offset (aperture rules)
  const unsigned lds_off = (unsigned)(size_t)l;
  const unsigned long long ga = (unsigned long long)(size_t)g;
  asm volatile("global_load_async_to_lds_b128 %0, %1, off"
               :: "v"(lds_off), "v"(ga) : "memory");
}

// ---------------------------------------------------------------- KNN (K=8)
__global__ __launch_bounds__(256)
void knn_kernel(const float* __restrict__ pos, int* __restrict__ idx, int N) {
  const int b = blockIdx.y;
  const int i = blockIdx.x * 256 + threadIdx.x;
  const float* pb = pos + (size_t)b * N * 2;
  const float xi = pb[i * 2 + 0];
  const float yi = pb[i * 2 + 1];
  float bd[8]; int bi[8];
#pragma unroll
  for (int k = 0; k < 8; k++) { bd[k] = 3.4e38f; bi[k] = 0; }
  __shared__ float sx[256], sy[256];
  for (int j0 = 0; j0 < N; j0 += 256) {
    __syncthreads();
    sx[threadIdx.x] = pb[(j0 + threadIdx.x) * 2 + 0];
    sy[threadIdx.x] = pb[(j0 + threadIdx.x) * 2 + 1];
    __syncthreads();
    for (int jj = 0; jj < 256; jj++) {
      const int j = j0 + jj;
      const float dx = xi - sx[jj];
      const float dy = yi - sy[jj];
      const float d = dx * dx + dy * dy;
      if (j == i) continue;
      if (d < bd[7]) {
        int p = 7;
        while (p > 0 && bd[p - 1] > d) { bd[p] = bd[p - 1]; bi[p] = bi[p - 1]; p--; }
        bd[p] = d; bi[p] = j;
      }
    }
  }
  int* o = idx + ((size_t)b * N + i) * 8;
#pragma unroll
  for (int k = 0; k < 8; k++) o[k] = bi[k];
}

// ------------------------------------------- weight convert f32 -> bf16, transposed
__global__ __launch_bounds__(256)
void convT_kernel(const float* __restrict__ W, __bf16* __restrict__ Wt, int K, int Nout) {
  const int t = blockIdx.x * 256 + threadIdx.x;
  if (t >= K * Nout) return;
  const int k = t / Nout;
  const int n = t - k * Nout;
  Wt[(size_t)n * K + k] = (__bf16)W[t];
}

// --------------------------------------- GIN gather: h = x_i + sum_{j in N(i)} x_j
template<int FD, typename TI>
__global__ __launch_bounds__(256)
void gather_sum(const TI* __restrict__ X, const int* __restrict__ idx,
                __bf16* __restrict__ out, int N) {
  const int b = blockIdx.y;
  const int node = blockIdx.x;
  const TI* xb = X + (size_t)b * N * FD;
  const int* id = idx + ((size_t)b * N + node) * 8;
  int j[8];
#pragma unroll
  for (int k = 0; k < 8; k++) j[k] = id[k];
  for (int f = threadIdx.x; f < FD; f += 256) {
    float s = (float)xb[(size_t)node * FD + f];
#pragma unroll
    for (int k = 0; k < 8; k++) s += (float)xb[(size_t)j[k] * FD + f];
    out[((size_t)b * N + node) * FD + f] = (__bf16)s;
  }
}

// --------------------------------------------------- bf16 WMMA GEMM + bias + act
// C[M x 256] = act(A[M x KTOT] @ Wt^T + bias),  Wt stored [256][KTOT] (transposed W)
// Double-buffered LDS; tiles staged with GLOBAL_LOAD_ASYNC_TO_LDS_B128 so the DMA
// of tile k+1 overlaps the 8 WMMAs of tile k.  ACT: 0 = relu, 1 = tanh, 2 = sigmoid
template<int KTOT, int ACT>
__global__ __launch_bounds__(256)
void gemm_bf16(const __bf16* __restrict__ A, const __bf16* __restrict__ Wt,
               const float* __restrict__ bias, __bf16* __restrict__ C, int M) {
  constexpr int BM = 128, BN = 128, BK = 32, LDT = BK + 8;  // 80B row: 16B-aligned, conflict-free
  constexpr int NT = KTOT / BK;
  __shared__ __align__(16) __bf16 As[2][BM * LDT];
  __shared__ __align__(16) __bf16 Bs[2][BN * LDT];

  const int tid  = threadIdx.x;
  const int wave = tid >> 5;          // 0..7
  const int lane = tid & 31;
  const int wm = wave & 1;            // 2 wave rows  (64 M-rows each)
  const int wn = wave >> 1;           // 4 wave cols  (32 N-cols each)
  const int mBase0 = blockIdx.x * BM;
  const int nBase0 = blockIdx.y * BN;

  const int kh = lane >> 4;           // half-wave select
  const int ln = lane & 15;

  f32x8 acc[4][2] = {};

  const int lrow = tid >> 1;          // 0..127 (tile row for staging)
  const int lcol = (tid & 1) * 16;    // 0 or 16
  const __bf16* gA = A  + (size_t)(mBase0 + lrow) * KTOT + lcol;
  const __bf16* gW = Wt + (size_t)(nBase0 + lrow) * KTOT + lcol;

  auto stage = [&](int buf, int k0) {
    async_copy_b128(gA + k0,     &As[buf][lrow * LDT + lcol]);
    async_copy_b128(gA + k0 + 8, &As[buf][lrow * LDT + lcol + 8]);
    async_copy_b128(gW + k0,     &Bs[buf][lrow * LDT + lcol]);
    async_copy_b128(gW + k0 + 8, &Bs[buf][lrow * LDT + lcol + 8]);
  };

  stage(0, 0);                         // prologue: fill buffer 0

  for (int it = 0; it < NT; ++it) {
    const int cur = it & 1;
    if (it + 1 < NT) {
      stage(cur ^ 1, (it + 1) * BK);   // kick off next tile's DMA
      asm volatile("s_wait_asynccnt 4" ::: "memory");  // only the 4 new ops may be in flight
    } else {
      asm volatile("s_wait_asynccnt 0" ::: "memory");
    }
    __syncthreads();                   // all waves' async writes into buf `cur` visible

    // A fragments: lane<16 -> row m, K{0..7,16..23}; lane>=16 -> K{8..15,24..31}
    bf16x16 afrag[4];
#pragma unroll
    for (int i = 0; i < 4; i++) {
      const int row = wm * 64 + i * 16 + ln;
      const int kk = kh * 8;
      *((bf16x8*)&afrag[i])     = *(const bf16x8*)&As[cur][row * LDT + kk];
      *((bf16x8*)&afrag[i] + 1) = *(const bf16x8*)&As[cur][row * LDT + kk + 16];
    }
    // B fragments from transposed weights: lane = column, K{0..7,16..23} / K{8..15,24..31}
    bf16x16 bfrag[2];
#pragma unroll
    for (int j = 0; j < 2; j++) {
      const int rn = wn * 32 + j * 16 + ln;
      const int kk = kh * 8;
      *((bf16x8*)&bfrag[j])     = *(const bf16x8*)&Bs[cur][rn * LDT + kk];
      *((bf16x8*)&bfrag[j] + 1) = *(const bf16x8*)&Bs[cur][rn * LDT + kk + 16];
    }
#pragma unroll
    for (int i = 0; i < 4; i++)
#pragma unroll
      for (int j = 0; j < 2; j++)
        acc[i][j] = __builtin_amdgcn_wmma_f32_16x16x32_bf16(
            false, afrag[i], false, bfrag[j], (short)0, acc[i][j], false, false);
    __syncthreads();                   // buf `cur` free for restaging at iter it+2
  }

  // epilogue: VGPR r -> M = r + 8*kh, N = ln  (16x16 f32 C layout)
#pragma unroll
  for (int i = 0; i < 4; i++) {
#pragma unroll
    for (int j = 0; j < 2; j++) {
      const int ncol = nBase0 + wn * 32 + j * 16 + ln;
      const float bv = bias[ncol];
#pragma unroll
      for (int r = 0; r < 8; r++) {
        const int mrow = mBase0 + wm * 64 + i * 16 + kh * 8 + r;
        float v = acc[i][j][r] + bv;
        if (ACT == 0)      v = v > 0.f ? v : 0.f;
        else if (ACT == 1) v = tanhf(v);
        else               v = 1.f / (1.f + __expf(-v));
        C[(size_t)mrow * 256 + ncol] = (__bf16)v;
      }
    }
  }
}

// ----------------------------------------------- attention score c = (a*b)@wc + cb
__global__ __launch_bounds__(256)
void score_kernel(const __bf16* __restrict__ ga, const __bf16* __restrict__ gb,
                  const float* __restrict__ wc, const float* __restrict__ cb,
                  float* __restrict__ c, int total) {
  const int node = blockIdx.x * 8 + (threadIdx.x >> 5);
  const int lane = threadIdx.x & 31;
  if (node >= total) return;
  const __bf16* pa = ga + (size_t)node * 256;
  const __bf16* pb = gb + (size_t)node * 256;
  float s = 0.f;
  for (int f = lane; f < 256; f += 32)
    s += (float)pa[f] * (float)pb[f] * wc[f];
#pragma unroll
  for (int off = 16; off > 0; off >>= 1) s += __shfl_down(s, off, 32);
  if (lane == 0) c[node] = s + cb[0];
}

// ------------------------------------------------- softmax over N (per batch, in place)
__global__ __launch_bounds__(256)
void softmax_kernel(float* __restrict__ c, int N) {
  float* cb = c + (size_t)blockIdx.x * N;
  __shared__ float red[256];
  float m = -3.4e38f;
  for (int n = threadIdx.x; n < N; n += 256) m = fmaxf(m, cb[n]);
  red[threadIdx.x] = m; __syncthreads();
  for (int s = 128; s > 0; s >>= 1) {
    if (threadIdx.x < s) red[threadIdx.x] = fmaxf(red[threadIdx.x], red[threadIdx.x + s]);
    __syncthreads();
  }
  m = red[0]; __syncthreads();
  float sum = 0.f;
  for (int n = threadIdx.x; n < N; n += 256) { const float e = __expf(cb[n] - m); cb[n] = e; sum += e; }
  red[threadIdx.x] = sum; __syncthreads();
  for (int s = 128; s > 0; s >>= 1) {
    if (threadIdx.x < s) red[threadIdx.x] += red[threadIdx.x + s];
    __syncthreads();
  }
  const float inv = 1.f / red[0];
  for (int n = threadIdx.x; n < N; n += 256) cb[n] *= inv;
}

// ----------------------------------- attention pooling h = sum attn*x3 (partials)
__global__ __launch_bounds__(256)
void pool_partial_kernel(const float* __restrict__ attn, const __bf16* __restrict__ x3,
                         float* __restrict__ partial, int N) {
  const int b = blockIdx.x;
  const int c = blockIdx.y;            // 32 chunks of N/32 rows
  const int f = threadIdx.x;
  const int span = N / 32;
  const int n0 = c * span;
  const float* ab = attn + (size_t)b * N;
  const __bf16* xb = x3 + (size_t)b * N * 256;
  float s = 0.f;
  for (int n = n0; n < n0 + span; n++) s += ab[n] * (float)xb[(size_t)n * 256 + f];
  partial[((size_t)b * 32 + c) * 256 + f] = s;
}

__global__ __launch_bounds__(256)
void pool_reduce_kernel(const float* __restrict__ partial, float* __restrict__ h) {
  const int b = blockIdx.x;
  const int f = threadIdx.x;
  float s = 0.f;
#pragma unroll
  for (int c = 0; c < 32; c++) s += partial[((size_t)b * 32 + c) * 256 + f];
  h[b * 256 + f] = s;
}

// ------------------------------------------------- rho + classifier head (tiny, f32)
__global__ __launch_bounds__(256)
void head_kernel(const float* __restrict__ h, const float* __restrict__ rho_w,
                 const float* __restrict__ rho_b, const float* __restrict__ cls_w,
                 const float* __restrict__ cls_b, float* __restrict__ out) {
  __shared__ float hr[4][256];
  const int o = threadIdx.x;
  for (int b = 0; b < 4; b++) {
    float s = rho_b[o];
    for (int f = 0; f < 256; f++) s += h[b * 256 + f] * rho_w[f * 256 + o];
    hr[b][o] = s > 0.f ? s : 0.f;
  }
  __syncthreads();
  if (o < 8) {
    const int b = o >> 1, cc = o & 1;
    float s = cls_b[cc];
    for (int f = 0; f < 256; f++) s += hr[b][f] * cls_w[f * 2 + cc];
    out[b * 2 + cc] = s;
  }
}

extern "C" void kernel_launch(void* const* d_in, const int* in_sizes, int n_in,
                              void* d_out, int out_size, void* d_ws, size_t ws_size,
                              hipStream_t stream) {
  constexpr int B = 4, N = 8192, F = 1024, H = 256;
  constexpr size_t M = (size_t)B * N;  // 32768

  const float* x      = (const float*)d_in[0];
  const float* pos    = (const float*)d_in[1];
  const float* w1_1   = (const float*)d_in[2];  const float* b1_1 = (const float*)d_in[3];
  const float* w1_2   = (const float*)d_in[4];  const float* b1_2 = (const float*)d_in[5];
  const float* w2_1   = (const float*)d_in[6];  const float* b2_1 = (const float*)d_in[7];
  const float* w2_2   = (const float*)d_in[8];  const float* b2_2 = (const float*)d_in[9];
  const float* w3_1   = (const float*)d_in[10]; const float* b3_1 = (const float*)d_in[11];
  const float* w3_2   = (const float*)d_in[12]; const float* b3_2 = (const float*)d_in[13];
  const float* att_aw = (const float*)d_in[14]; const float* att_ab = (const float*)d_in[15];
  const float* att_bw = (const float*)d_in[16]; const float* att_bb = (const float*)d_in[17];
  const float* att_cw = (const float*)d_in[18]; const float* att_cb = (const float*)d_in[19];
  const float* rho_w  = (const float*)d_in[20]; const float* rho_b  = (const float*)d_in[21];
  const float* cls_w  = (const float*)d_in[22]; const float* cls_b  = (const float*)d_in[23];

  char* w = (char*)d_ws;
  size_t off = 0;
  auto take = [&](size_t bytes) -> void* {
    void* p = w + off;
    off = (off + bytes + 255) & ~(size_t)255;
    return p;
  };
  int*    idx     = (int*)take(M * 8 * sizeof(int));
  __bf16* wt11    = (__bf16*)take((size_t)F * H * 2);
  __bf16* wt12    = (__bf16*)take((size_t)H * H * 2);
  __bf16* wt21    = (__bf16*)take((size_t)H * H * 2);
  __bf16* wt22    = (__bf16*)take((size_t)H * H * 2);
  __bf16* wt31    = (__bf16*)take((size_t)H * H * 2);
  __bf16* wt32    = (__bf16*)take((size_t)H * H * 2);
  __bf16* wta     = (__bf16*)take((size_t)H * H * 2);
  __bf16* wtb     = (__bf16*)take((size_t)H * H * 2);
  float*  scores  = (float*)take(M * sizeof(float));
  float*  partial = (float*)take((size_t)B * 32 * H * sizeof(float));
  float*  pooled  = (float*)take((size_t)B * H * sizeof(float));
  __bf16* g1      = (__bf16*)take(M * F * 2);   // 64 MB
  __bf16* bufA    = (__bf16*)take(M * H * 2);   // 16 MB
  __bf16* bufB    = (__bf16*)take(M * H * 2);
  __bf16* bufC    = (__bf16*)take(M * H * 2);
  (void)ws_size; (void)in_sizes; (void)n_in; (void)out_size;

  const dim3 gGemm(M / 128, 2);
  const dim3 gGather(N, B);

  // KNN graph
  knn_kernel<<<dim3(N / 256, B), 256, 0, stream>>>(pos, idx, N);
  // weights -> bf16, transposed [Nout][K]
  convT_kernel<<<(F * H + 255) / 256, 256, 0, stream>>>(w1_1, wt11, F, H);
  convT_kernel<<<(H * H + 255) / 256, 256, 0, stream>>>(w1_2, wt12, H, H);
  convT_kernel<<<(H * H + 255) / 256, 256, 0, stream>>>(w2_1, wt21, H, H);
  convT_kernel<<<(H * H + 255) / 256, 256, 0, stream>>>(w2_2, wt22, H, H);
  convT_kernel<<<(H * H + 255) / 256, 256, 0, stream>>>(w3_1, wt31, H, H);
  convT_kernel<<<(H * H + 255) / 256, 256, 0, stream>>>(w3_2, wt32, H, H);
  convT_kernel<<<(H * H + 255) / 256, 256, 0, stream>>>(att_aw, wta, H, H);
  convT_kernel<<<(H * H + 255) / 256, 256, 0, stream>>>(att_bw, wtb, H, H);

  // GIN layer 1 (F -> H -> H)
  gather_sum<F, float><<<gGather, 256, 0, stream>>>(x, idx, g1, N);
  gemm_bf16<F, 0><<<gGemm, 256, 0, stream>>>(g1, wt11, b1_1, bufA, (int)M);
  gemm_bf16<H, 0><<<gGemm, 256, 0, stream>>>(bufA, wt12, b1_2, bufB, (int)M);  // x1
  // GIN layer 2
  gather_sum<H, __bf16><<<gGather, 256, 0, stream>>>(bufB, idx, bufC, N);
  gemm_bf16<H, 0><<<gGemm, 256, 0, stream>>>(bufC, wt21, b2_1, bufA, (int)M);
  gemm_bf16<H, 0><<<gGemm, 256, 0, stream>>>(bufA, wt22, b2_2, bufB, (int)M);  // x2
  // GIN layer 3
  gather_sum<H, __bf16><<<gGather, 256, 0, stream>>>(bufB, idx, bufC, N);
  gemm_bf16<H, 0><<<gGemm, 256, 0, stream>>>(bufC, wt31, b3_1, bufA, (int)M);
  gemm_bf16<H, 0><<<gGemm, 256, 0, stream>>>(bufA, wt32, b3_2, bufB, (int)M);  // x3

  // gated attention: a = tanh(x3@Wa+ba), b = sigmoid(x3@Wb+bb)
  gemm_bf16<H, 1><<<gGemm, 256, 0, stream>>>(bufB, wta, att_ab, bufA, (int)M);
  gemm_bf16<H, 2><<<gGemm, 256, 0, stream>>>(bufB, wtb, att_bb, bufC, (int)M);
  score_kernel<<<M / 8, 256, 0, stream>>>(bufA, bufC, att_cw, att_cb, scores, (int)M);
  softmax_kernel<<<B, 256, 0, stream>>>(scores, N);
  pool_partial_kernel<<<dim3(B, 32), 256, 0, stream>>>(scores, bufB, partial, N);
  pool_reduce_kernel<<<B, 256, 0, stream>>>(partial, pooled);
  head_kernel<<<1, 256, 0, stream>>>(pooled, rho_w, rho_b, cls_w, cls_b, (float*)d_out);
}
